// HausdorffLoss_19112604467902
// MI455X (gfx1250) — compile-verified, matrix operand order checked by reading, not run on hardware
//
#include <hip/hip_runtime.h>
#include <stdint.h>

// Problem geometry (fixed by reference): B=2, C=4, D=H=W=128
#define VOLSZ   (128*128*128)      // voxels per (vol,b) = 2097152 = 1<<21
#define NVOX    (2*VOLSZ)          // 4194304 voxels per binary volume set
#define WORDS_PER_BVOL (VOLSZ/32)  // 65536 packed u32 words per (vol,b)
#define ROWS_PER_BVOL  (128*128)   // 16384 uint4 rows per (vol,b)
#define NBVOL   12                 // 6 binary volumes (3 target, 3 prob) x B=2
#define MAX_STEPS 29
#define TD 16                      // erode tile: d extent
#define THH 16                     // erode tile: h extent

typedef __attribute__((ext_vector_type(16))) _Float16 v16h;
typedef __attribute__((ext_vector_type(8)))  float    v8f;
typedef int a4i __attribute__((vector_size(16)));   // matches builtin param type

// ---- CDNA5 async global->LDS staging (guarded; fallback = ds_store) -------
#if defined(__has_builtin)
#  if __has_builtin(__builtin_amdgcn_global_load_async_to_lds_b128)
#    define HAVE_ASYNC_LDS 1
#  endif
#  if __has_builtin(__builtin_amdgcn_s_wait_asynccnt)
#    define HAVE_ASYNC_WAIT 1
#  endif
#endif

__device__ __forceinline__ void async_copy16(const uint4* g, uint4* l) {
#if defined(HAVE_ASYNC_LDS)
    // global_load_async_to_lds_b128: ASYNCcnt-tracked, bypasses VGPRs.
    // Low 32 bits of a flat LDS pointer are the wave-relative LDS offset.
    __builtin_amdgcn_global_load_async_to_lds_b128(
        (__attribute__((address_space(1))) a4i*)(uintptr_t)g,
        (__attribute__((address_space(3))) a4i*)(uintptr_t)l,
        0, 0);
#else
    *l = *g;
#endif
}

__device__ __forceinline__ void async_join() {
#if defined(HAVE_ASYNC_LDS)
#  if defined(HAVE_ASYNC_WAIT)
    __builtin_amdgcn_s_wait_asynccnt(0);
#  else
    asm volatile("s_wait_asynccnt 0x0" ::: "memory");
#  endif
#endif
}

// ---------------------------------------------------------------------------
// Kernel 1: softmax over C=4, pack the 6 "neg" binary volumes as bitmasks.
// bit = 1 means background (field to be eroded): negT_c = (target != c),
// negP_c = !(p_c > 0.5). wave32: 32 lanes == 32 consecutive W voxels.
// ---------------------------------------------------------------------------
__global__ __launch_bounds__(256)
void hd_pack_kernel(const float* __restrict__ logits,
                    const int*   __restrict__ target,
                    uint32_t*    __restrict__ pack) {
    const int n   = blockIdx.x * blockDim.x + threadIdx.x;   // voxel id
    const int b   = n >> 21;
    const int rem = n & (VOLSZ - 1);
    const int t   = target[n];

    const float* lb = logits + (size_t)b * 4 * VOLSZ + rem;
    const float l0 = lb[0];
    const float l1 = lb[VOLSZ];
    const float l2 = lb[2 * VOLSZ];
    const float l3 = lb[3 * VOLSZ];
    const float m  = fmaxf(fmaxf(l0, l1), fmaxf(l2, l3));
    const float e0 = __expf(l0 - m);
    const float e1 = __expf(l1 - m);
    const float e2 = __expf(l2 - m);
    const float e3 = __expf(l3 - m);
    const float inv = 1.0f / (e0 + e1 + e2 + e3);
    const float p1 = e1 * inv, p2 = e2 * inv, p3 = e3 * inv;

    const unsigned mT1 = (unsigned)__ballot(t != 1);
    const unsigned mT2 = (unsigned)__ballot(t != 2);
    const unsigned mT3 = (unsigned)__ballot(t != 3);
    const unsigned mP1 = (unsigned)__ballot(!(p1 > 0.5f));
    const unsigned mP2 = (unsigned)__ballot(!(p2 > 0.5f));
    const unsigned mP3 = (unsigned)__ballot(!(p3 > 0.5f));

    if ((threadIdx.x & 31) == 0) {
        const int lw = (n >> 5) & (WORDS_PER_BVOL - 1);  // local word in bvol
        pack[(0 * 2 + b) * WORDS_PER_BVOL + lw] = mT1;
        pack[(1 * 2 + b) * WORDS_PER_BVOL + lw] = mT2;
        pack[(2 * 2 + b) * WORDS_PER_BVOL + lw] = mT3;
        pack[(3 * 2 + b) * WORDS_PER_BVOL + lw] = mP1;
        pack[(4 * 2 + b) * WORDS_PER_BVOL + lw] = mP2;
        pack[(5 * 2 + b) * WORDS_PER_BVOL + lw] = mP3;
    }
}

// ---------------------------------------------------------------------------
// Bit-packed erosion along W of one 128-bit row (out-of-range = 1 / AND-id).
// bit i of word j  <->  voxel w = j*32 + i
// ---------------------------------------------------------------------------
__device__ __forceinline__ uint4 erodeW(uint4 x) {
    uint4 e;
    const uint32_t l0 = (x.x << 1) | 1u;            // w = -1 border
    const uint32_t l1 = (x.y << 1) | (x.x >> 31);
    const uint32_t l2 = (x.z << 1) | (x.y >> 31);
    const uint32_t l3 = (x.w << 1) | (x.z >> 31);
    const uint32_t r0 = (x.x >> 1) | (x.y << 31);
    const uint32_t r1 = (x.y >> 1) | (x.z << 31);
    const uint32_t r2 = (x.z >> 1) | (x.w << 31);
    const uint32_t r3 = (x.w >> 1) | 0x80000000u;   // w = 128 border
    e.x = x.x & l0 & r0;
    e.y = x.y & l1 & r1;
    e.z = x.z & l2 & r2;
    e.w = x.w & l3 & r3;
    return e;
}

__device__ __forceinline__ uint4 and4(uint4 a, uint4 b) {
    return make_uint4(a.x & b.x, a.y & b.y, a.z & b.z, a.w & b.w);
}

// ---------------------------------------------------------------------------
// Kernel 2: one erosion step (27-neighborhood AND) on all 12 packed volumes.
// Block = 16x16 rows of one bvol; 18x18 halo tile staged into LDS with
// async global->LDS loads. One thread per 128-voxel W row; flipped voxels
// record the step number into the u8 distance map.
// ---------------------------------------------------------------------------
__global__ __launch_bounds__(256)
void hd_erode_kernel(const uint4* __restrict__ cur4,
                     uint4*       __restrict__ nxt4,
                     uint8_t*     __restrict__ dist,
                     int step) {
    __shared__ uint4 tile[(TD + 2) * (THH + 2)];     // 18*18*16B = 5184 B

    const int bvol = blockIdx.z;                     // 0..11
    const int d0   = blockIdx.y * TD;
    const int h0   = blockIdx.x * THH;
    const int tid  = threadIdx.x;
    const int rowBase = bvol << 14;

    // Stage halo tile: in-range rows via async loads, border rows = all-ones.
    for (int slot = tid; slot < (TD + 2) * (THH + 2); slot += 256) {
        const int ld = slot / (THH + 2);
        const int lh = slot - ld * (THH + 2);
        const int d  = d0 + ld - 1;
        const int h  = h0 + lh - 1;
        if ((unsigned)d < 128u && (unsigned)h < 128u) {
            async_copy16(&cur4[rowBase + (d << 7) + h], &tile[slot]);
        } else {
            tile[slot] = make_uint4(~0u, ~0u, ~0u, ~0u);
        }
    }
    async_join();          // drain this wave's ASYNCcnt before the barrier
    __syncthreads();

    // Compute one output row per thread.
    const int ld = tid >> 4;
    const int lh = tid & 15;
    const int d  = d0 + ld;
    const int h  = h0 + lh;

    const uint4 center = tile[(ld + 1) * (THH + 2) + (lh + 1)];
    uint4 e = make_uint4(~0u, ~0u, ~0u, ~0u);
    #pragma unroll
    for (int dd = 0; dd < 3; ++dd) {
        #pragma unroll
        for (int dh = 0; dh < 3; ++dh) {
            e = and4(e, erodeW(tile[(ld + dd) * (THH + 2) + (lh + dh)]));
        }
    }

    const int r = rowBase + (d << 7) + h;
    nxt4[r] = e;

    // voxels that just eroded away: distance = step
    uint4 ch;
    ch.x = center.x & ~e.x;
    ch.y = center.y & ~e.y;
    ch.z = center.z & ~e.z;
    ch.w = center.w & ~e.w;

    uint8_t* dbase = dist + ((size_t)bvol << 21) + ((size_t)((d << 7) + h) << 7);
    unsigned mm;
    mm = ch.x; while (mm) { int i = __ffs(mm) - 1; mm &= mm - 1; dbase[i]      = (uint8_t)step; }
    mm = ch.y; while (mm) { int i = __ffs(mm) - 1; mm &= mm - 1; dbase[32 + i] = (uint8_t)step; }
    mm = ch.z; while (mm) { int i = __ffs(mm) - 1; mm &= mm - 1; dbase[64 + i] = (uint8_t)step; }
    mm = ch.w; while (mm) { int i = __ffs(mm) - 1; mm &= mm - 1; dbase[96 + i] = (uint8_t)step; }
}

// ---------------------------------------------------------------------------
// Kernel 3: per-voxel loss, block reduction via WMMA (16x16x32 f16 -> f32
// accumulate against an all-ones B matrix), one f32 atomic per block.
// ---------------------------------------------------------------------------
__global__ __launch_bounds__(256)
void hd_loss_kernel(const float*  __restrict__ logits,
                    const int*    __restrict__ target,
                    const uint8_t* __restrict__ dist,
                    float*        __restrict__ acc) {
    const int n   = blockIdx.x * blockDim.x + threadIdx.x;
    const int b   = n >> 21;
    const int rem = n & (VOLSZ - 1);
    const int t   = target[n];

    const float* lb = logits + (size_t)b * 4 * VOLSZ + rem;
    const float l0 = lb[0];
    const float l1 = lb[VOLSZ];
    const float l2 = lb[2 * VOLSZ];
    const float l3 = lb[3 * VOLSZ];
    const float m  = fmaxf(fmaxf(l0, l1), fmaxf(l2, l3));
    const float e0 = __expf(l0 - m);
    const float e1 = __expf(l1 - m);
    const float e2 = __expf(l2 - m);
    const float e3 = __expf(l3 - m);
    const float inv = 1.0f / (e0 + e1 + e2 + e3);
    const float p[3] = { e1 * inv, e2 * inv, e3 * inv };

    float partial = 0.0f;
    #pragma unroll
    for (int c = 1; c <= 3; ++c) {
        const float tc  = (t == c) ? 1.0f : 0.0f;
        const float dtv = (float)dist[((size_t)((c - 1) * 2 + b) << 21) + rem];
        const float dpv = (float)dist[((size_t)((3 + (c - 1)) * 2 + b) << 21) + rem];
        const float df  = p[c - 1] - tc;
        partial += df * df * (dtv * dtv + dpv * dpv);
    }

    __shared__ float red[256];
    red[threadIdx.x] = partial;
    __syncthreads();

    if (threadIdx.x < 32) {             // wave 0 only: EXEC all-ones in-wave
        v16h a;
        #pragma unroll
        for (int i = 0; i < 8; ++i)  a[i] = (_Float16)red[threadIdx.x * 8 + i];
        #pragma unroll
        for (int i = 8; i < 16; ++i) a[i] = (_Float16)0.0f;
        v16h ones;
        #pragma unroll
        for (int i = 0; i < 16; ++i) ones[i] = (_Float16)1.0f;
        v8f cacc = {0.f, 0.f, 0.f, 0.f, 0.f, 0.f, 0.f, 0.f};
        // D = A x Ones + C: every D column holds the 16 row-sums of A.
        cacc = __builtin_amdgcn_wmma_f32_16x16x32_f16(
            /*neg_a=*/false, a, /*neg_b=*/false, ones,
            /*c_mod=*/(short)0, cacc, /*reuse_a=*/false, /*reuse_b=*/false);
        float s = 0.0f;
        #pragma unroll
        for (int i = 0; i < 8; ++i) s += cacc[i];   // rows 0-7 (lanes<16) / 8-15
        s += __shfl_xor(s, 16, 32);                 // combine both row halves
        if (threadIdx.x == 0) atomicAdd(acc, s);
    }
}

__global__ void hd_finalize_kernel(const float* __restrict__ acc,
                                   float* __restrict__ out) {
    out[0] = acc[0] * (1.0f / (4194304.0f * 3.0f));  // mean over B*D*H*W, /(C-1)
}

// ---------------------------------------------------------------------------
extern "C" void kernel_launch(void* const* d_in, const int* in_sizes, int n_in,
                              void* d_out, int out_size, void* d_ws, size_t ws_size,
                              hipStream_t stream) {
    (void)in_sizes; (void)n_in; (void)out_size; (void)ws_size;
    const float* logits = (const float*)d_in[0];
    const int*   target = (const int*)d_in[1];

    uint8_t* ws = (uint8_t*)d_ws;
    const size_t packBytes = (size_t)NBVOL * WORDS_PER_BVOL * 4;  // 3,145,728
    const size_t distBytes = (size_t)NBVOL * VOLSZ;               // 25,165,824
    uint32_t* packA = (uint32_t*)(ws);
    uint32_t* packB = (uint32_t*)(ws + packBytes);
    uint8_t*  dist  = ws + 2 * packBytes;
    float*    acc   = (float*)(ws + 2 * packBytes + distBytes);

    (void)hipMemsetAsync(dist, 0, distBytes, stream); // unflipped voxels -> 0
    (void)hipMemsetAsync(acc, 0, sizeof(float), stream);

    hd_pack_kernel<<<NVOX / 256, 256, 0, stream>>>(logits, target, packA);

    uint32_t* cur = packA;
    uint32_t* nxt = packB;
    dim3 egrid(128 / THH, 128 / TD, NBVOL);           // 8 x 8 x 12 = 768 blocks
    for (int s = 1; s <= MAX_STEPS; ++s) {
        hd_erode_kernel<<<egrid, 256, 0, stream>>>(
            (const uint4*)cur, (uint4*)nxt, dist, s);
        uint32_t* tmp = cur; cur = nxt; nxt = tmp;
    }

    hd_loss_kernel<<<NVOX / 256, 256, 0, stream>>>(logits, target, dist, acc);
    hd_finalize_kernel<<<1, 1, 0, stream>>>(acc, (float*)d_out);
}
